// CosineSimilarity_74259984548271
// MI455X (gfx1250) — matrix operand order repaired and checked by reading, not compiled, not run on hardware
//
#include <hip/hip_runtime.h>
#include <hip/hip_bf16.h>
#include <cstddef>

#define WAYS   5
#define SHOTS  5
#define NSAMP  26
#define FDIM   1024
#define HID    512
#define NBATCH 2048
#define BN_EPS 1e-5f

typedef __attribute__((ext_vector_type(16))) __bf16 v16bf;
typedef __attribute__((ext_vector_type(8)))  __bf16 v8bf;
typedef __attribute__((ext_vector_type(8)))  float  v8f;

// ---------------- LDS layout (bytes), bf16 rows with padded strides --------
#define NF_STRIDE 1040        // 1024 + 16-elem pad (16B aligned rows)
#define X_STRIDE  528         // 512 + 16
#define OFF_NF 0              // 26 x 1040 bf16 = 54080 B
#define OFF_X1 54080          // 26 x 528  bf16 = 27456 B
#define OFF_X2 81536          // 26 x 528  bf16
#define OFF_P  108992         // 26 x 1040 bf16
#define OFF_Q  163072         // 26 x 1040 bf16
#define OFF_A2 217152         // 26 x 1040 bf16 (ends 271232)
#define SMEM_BYTES 294912     // + guard for padded-row (26..31) OOB reads
#define OFF_DOTS OFF_X1       // 32x32 f32 (x1 dead by then)
#define OFF_N2   (OFF_X1 + 4096)

// ---------------- workspace layout (bytes) --------------------------------
#define WS_W1B 0              // 512*1024 bf16 = 1,048,576
#define WS_W2B 1048576        // 512*512  bf16 =   524,288
#define WS_W3B 1572864        // 1024*512 bf16 = 1,048,576
#define WS_B1F 2621440        // 512 f32
#define WS_B2F 2623488        // 512 f32
#define WS_B3F 2625536        // 1024 f32

// ===========================================================================
// Prep: fold BN scale into conv weights, convert to bf16, fold BN bias.
// ===========================================================================
__global__ void fold_weights_kernel(
    const float* __restrict__ w1, const float* __restrict__ b1,
    const float* __restrict__ g1, const float* __restrict__ be1,
    const float* __restrict__ m1, const float* __restrict__ v1,
    const float* __restrict__ w2, const float* __restrict__ b2,
    const float* __restrict__ g2, const float* __restrict__ be2,
    const float* __restrict__ m2, const float* __restrict__ v2,
    const float* __restrict__ w3, const float* __restrict__ b3,
    __bf16* __restrict__ w1b, __bf16* __restrict__ w2b, __bf16* __restrict__ w3b,
    float* __restrict__ b1f, float* __restrict__ b2f, float* __restrict__ b3f)
{
    int idx = blockIdx.x * 256 + threadIdx.x;
    if (idx < HID * FDIM) {                       // w1' = s1 * w1
        int o = idx >> 10;
        float s = g1[o] * rsqrtf(v1[o] + BN_EPS);
        w1b[idx] = (__bf16)(w1[idx] * s);
        return;
    }
    idx -= HID * FDIM;
    if (idx < HID * HID) {                        // w2' = s2 * w2
        int o = idx >> 9;
        float s = g2[o] * rsqrtf(v2[o] + BN_EPS);
        w2b[idx] = (__bf16)(w2[idx] * s);
        return;
    }
    idx -= HID * HID;
    if (idx < FDIM * HID) {                       // w3 straight convert
        w3b[idx] = (__bf16)w3[idx];
        return;
    }
    idx -= FDIM * HID;
    if (idx < HID) {                              // b1' = s1*(b1-m1)+be1
        float s = g1[idx] * rsqrtf(v1[idx] + BN_EPS);
        b1f[idx] = (b1[idx] - m1[idx]) * s + be1[idx];
        return;
    }
    idx -= HID;
    if (idx < HID) {
        float s = g2[idx] * rsqrtf(v2[idx] + BN_EPS);
        b2f[idx] = (b2[idx] - m2[idx]) * s + be2[idx];
        return;
    }
    idx -= HID;
    if (idx < FDIM) b3f[idx] = b3[idx];
}
#define PREP_ELEMS (HID*FDIM + HID*HID + FDIM*HID + HID + HID + FDIM)

// ===========================================================================
// WMMA fragment helpers (bf16 16x16x32). Layouts per CDNA5 ISA 7.12.2:
//  A: lane holds row M=lane&15; K chunks [kA,kA+8) and [kA+16,kA+24),
//     kA = k0 + (lane>=16 ? 8 : 0).
//  B: lane holds column N=lane&15; K chunks [kB,kB+8) and [kB+16,kB+24),
//     kB = k0 + (lane>=16 ? 8 : 0)  -> a row of row-major [N x K] is K-contig.
//  C/D: lane gives N=lane&15; VGPR i gives M = i + (lane>=16 ? 8 : 0).
// ===========================================================================
__device__ __forceinline__ v16bf load_a_frag(const __bf16* base, int m0, int k0,
                                             int stride, int lane)
{
    int r  = m0 + (lane & 15);
    int kA = k0 + ((lane >> 4) << 3);
    const __bf16* p = base + (size_t)r * stride + kA;
    union { v16bf v; v8bf h[2]; } u;
    u.h[0] = *reinterpret_cast<const v8bf*>(p);
    u.h[1] = *reinterpret_cast<const v8bf*>(p + 16);
    return u.v;
}

__device__ __forceinline__ v16bf load_b_frag(const __bf16* base, int n0, int k0,
                                             int stride, int lane)
{
    int n = n0 + (lane & 15);
    int k = k0 + ((lane >> 4) << 3);
    const __bf16* p = base + (size_t)n * stride + k;
    union { v16bf v; v8bf h[2]; } u;
    u.h[0] = *reinterpret_cast<const v8bf*>(p);
    u.h[1] = *reinterpret_cast<const v8bf*>(p + 16);
    return u.v;
}

// Single-tile GEMM (used for the small 26x26 Gram matrices).
__device__ __forceinline__ v8f gemm_tile(const __bf16* A, int aStride, int m0,
                                         const __bf16* B, int bStride, int n0,
                                         int K, int lane)
{
    v8f acc = {0.f,0.f,0.f,0.f,0.f,0.f,0.f,0.f};
    for (int k0 = 0; k0 < K; k0 += 32) {
        v16bf af = load_a_frag(A, m0, k0, aStride, lane);
        v16bf bf = load_b_frag(B, n0, k0, bStride, lane);
        acc = __builtin_amdgcn_wmma_f32_16x16x32_bf16(
                  false, af, false, bf, (short)0, acc, false, false);
    }
    return acc;
}

// Dual-M GEMM: both M-tiles (m0=0,16) share one B (weight) fragment per
// k-step -> halves VMEM fetches per WMMA vs tile-at-a-time.
__device__ __forceinline__ void gemm_dualM(const __bf16* A, int aStride,
                                           const __bf16* B, int bStride, int n0,
                                           int K, int lane,
                                           v8f& acc0, v8f& acc1)
{
    acc0 = (v8f){0.f,0.f,0.f,0.f,0.f,0.f,0.f,0.f};
    acc1 = (v8f){0.f,0.f,0.f,0.f,0.f,0.f,0.f,0.f};
    for (int k0 = 0; k0 < K; k0 += 32) {
        v16bf bf  = load_b_frag(B, n0, k0, bStride, lane);
        v16bf af0 = load_a_frag(A, 0,  k0, aStride, lane);
        v16bf af1 = load_a_frag(A, 16, k0, aStride, lane);
        acc0 = __builtin_amdgcn_wmma_f32_16x16x32_bf16(
                   false, af0, false, bf, (short)0, acc0, false, false);
        acc1 = __builtin_amdgcn_wmma_f32_16x16x32_bf16(
                   false, af1, false, bf, (short)0, acc1, false, false);
    }
}

// ===========================================================================
// Main kernel: one workgroup (8 wave32) per episode.
// ===========================================================================
__global__ void episode_kernel(const float* __restrict__ node_feats,
                               const __bf16* __restrict__ w1b,
                               const __bf16* __restrict__ w2b,
                               const __bf16* __restrict__ w3b,
                               const float* __restrict__ b1f,
                               const float* __restrict__ b2f,
                               const float* __restrict__ b3f,
                               float* __restrict__ out)
{
    extern __shared__ char smem[];
    __bf16* nfL  = (__bf16*)(smem + OFF_NF);
    __bf16* x1L  = (__bf16*)(smem + OFF_X1);
    __bf16* x2L  = (__bf16*)(smem + OFF_X2);
    __bf16* pL   = (__bf16*)(smem + OFF_P);
    __bf16* qL   = (__bf16*)(smem + OFF_Q);
    __bf16* a2L  = (__bf16*)(smem + OFF_A2);

    const int tid  = threadIdx.x;
    const int lane = tid & 31;
    const int wave = tid >> 5;
    const int b    = blockIdx.x;

    // ---- Phase 1: shot-mean support rows, stage nf as bf16 in LDS --------
    {
        const float* src = node_feats + (size_t)b * NSAMP * FDIM;
        for (int f = tid; f < FDIM; f += 256) {
            #pragma unroll
            for (int w = 0; w < WAYS; ++w) {
                float s = 0.f;
                #pragma unroll
                for (int sh = 0; sh < SHOTS; ++sh)
                    s += src[(w * SHOTS + sh) * FDIM + f];
                __bf16 mv = (__bf16)(s * 0.2f);
                #pragma unroll
                for (int sh = 0; sh < SHOTS; ++sh)
                    nfL[(w * SHOTS + sh) * NF_STRIDE + f] = mv;
            }
            nfL[25 * NF_STRIDE + f] = (__bf16)src[25 * FDIM + f];
        }
    }
    __syncthreads();

    // ---- Phase 2: x1 = lrelu(nf @ w1'^T + b1')  [26 x 512] ----------------
    for (int nt = wave; nt < 32; nt += 8) {
        int n0 = nt * 16;
        v8f acc0, acc1;
        gemm_dualM(nfL, NF_STRIDE, w1b, FDIM, n0, FDIM, lane, acc0, acc1);
        int n = n0 + (lane & 15);
        float bias = b1f[n];
        int mh = (lane >> 4) << 3;
        #pragma unroll
        for (int i = 0; i < 8; ++i) {
            float d0 = acc0[i] + bias;
            d0 = d0 > 0.f ? d0 : 0.01f * d0;
            int m0 = mh + i;
            if (m0 < NSAMP) x1L[m0 * X_STRIDE + n] = (__bf16)d0;
            float d1 = acc1[i] + bias;
            d1 = d1 > 0.f ? d1 : 0.01f * d1;
            int m1 = 16 + mh + i;
            if (m1 < NSAMP) x1L[m1 * X_STRIDE + n] = (__bf16)d1;
        }
    }
    __syncthreads();

    // ---- Phase 3: x2 = lrelu(x1 @ w2'^T + b2')  [26 x 512] ----------------
    for (int nt = wave; nt < 32; nt += 8) {
        int n0 = nt * 16;
        v8f acc0, acc1;
        gemm_dualM(x1L, X_STRIDE, w2b, HID, n0, HID, lane, acc0, acc1);
        int n = n0 + (lane & 15);
        float bias = b2f[n];
        int mh = (lane >> 4) << 3;
        #pragma unroll
        for (int i = 0; i < 8; ++i) {
            float d0 = acc0[i] + bias;
            d0 = d0 > 0.f ? d0 : 0.01f * d0;
            int m0 = mh + i;
            if (m0 < NSAMP) x2L[m0 * X_STRIDE + n] = (__bf16)d0;
            float d1 = acc1[i] + bias;
            d1 = d1 > 0.f ? d1 : 0.01f * d1;
            int m1 = 16 + mh + i;
            if (m1 < NSAMP) x2L[m1 * X_STRIDE + n] = (__bf16)d1;
        }
    }
    __syncthreads();

    // ---- Phase 4: att = sigmoid(x2 @ w3^T + b3); emit a2, P, Q -----------
    for (int nt = wave; nt < 64; nt += 8) {
        int n0 = nt * 16;
        v8f acc0, acc1;
        gemm_dualM(x2L, X_STRIDE, w3b, HID, n0, HID, lane, acc0, acc1);
        int n = n0 + (lane & 15);                   // feature index f
        float bias = b3f[n];
        int mh = (lane >> 4) << 3;
        #pragma unroll
        for (int i = 0; i < 8; ++i) {
            #pragma unroll
            for (int half = 0; half < 2; ++half) {
                float d = (half ? acc1[i] : acc0[i]) + bias;
                float a = 1.0f / (1.0f + __expf(-d));   // sigmoid
                float a2 = a * a;
                int m = half * 16 + mh + i;
                if (m < NSAMP) {
                    float nfv = (float)nfL[m * NF_STRIDE + n];
                    float Pv  = a2 * nfv;
                    float Qv  = Pv * nfv;
                    a2L[m * NF_STRIDE + n] = (__bf16)a2;
                    pL [m * NF_STRIDE + n] = (__bf16)Pv;
                    qL [m * NF_STRIDE + n] = (__bf16)Qv;
                }
            }
        }
    }
    __syncthreads();

    // ---- Phase 5: dots = P @ P^T ; n2 = Q @ a2^T  (each 26x26, K=1024) ---
    {
        float* dotsL = (float*)(smem + OFF_DOTS);
        float* n2L   = (float*)(smem + OFF_N2);
        int g  = wave & 1;                           // 0: dots, 1: n2
        int m0 = ((wave >> 1) & 1) * 16;
        int n0 = ((wave >> 2) & 1) * 16;
        const __bf16* A = g ? qL : pL;
        const __bf16* B = g ? a2L : pL;              // row n of B == column n
        v8f acc = gemm_tile(A, NF_STRIDE, m0, B, NF_STRIDE, n0, FDIM, lane);
        float* dst = g ? n2L : dotsL;
        int n = n0 + (lane & 15);
        int mbase = m0 + ((lane >> 4) << 3);
        #pragma unroll
        for (int i = 0; i < 8; ++i)
            dst[(mbase + i) * 32 + n] = acc[i];
    }
    __syncthreads();

    // ---- Phase 6: sim = dots / (sqrt(n2_ij)*sqrt(n2_ji)); clip & store ---
    {
        const float* dotsL = (const float*)(smem + OFF_DOTS);
        const float* n2L   = (const float*)(smem + OFF_N2);
        size_t base = (size_t)b * 2 * NSAMP * NSAMP;
        for (int idx = tid; idx < NSAMP * NSAMP; idx += 256) {
            int i = idx / NSAMP, j = idx % NSAMP;
            float dd  = dotsL[i * 32 + j];
            float nij = n2L[i * 32 + j];
            float nji = n2L[j * 32 + i];
            float sim = dd * rsqrtf(nij * nji);
            float s0 = fminf(fmaxf(sim, 0.f), 1.f);
            float s1 = fminf(fmaxf(1.f - sim, 0.f), 1.f);
            out[base + idx]                 = s0;
            out[base + NSAMP * NSAMP + idx] = s1;
        }
    }
}

// ===========================================================================
extern "C" void kernel_launch(void* const* d_in, const int* in_sizes, int n_in,
                              void* d_out, int out_size, void* d_ws, size_t ws_size,
                              hipStream_t stream)
{
    (void)in_sizes; (void)n_in; (void)out_size; (void)ws_size;
    const float* node = (const float*)d_in[0];
    const float* w1  = (const float*)d_in[1];
    const float* b1  = (const float*)d_in[2];
    const float* g1  = (const float*)d_in[3];
    const float* be1 = (const float*)d_in[4];
    const float* m1  = (const float*)d_in[5];
    const float* v1  = (const float*)d_in[6];
    const float* w2  = (const float*)d_in[7];
    const float* b2  = (const float*)d_in[8];
    const float* g2  = (const float*)d_in[9];
    const float* be2 = (const float*)d_in[10];
    const float* m2  = (const float*)d_in[11];
    const float* v2  = (const float*)d_in[12];
    const float* w3  = (const float*)d_in[13];
    const float* b3  = (const float*)d_in[14];

    char* ws = (char*)d_ws;
    __bf16* w1b = (__bf16*)(ws + WS_W1B);
    __bf16* w2b = (__bf16*)(ws + WS_W2B);
    __bf16* w3b = (__bf16*)(ws + WS_W3B);
    float*  b1f = (float*)(ws + WS_B1F);
    float*  b2f = (float*)(ws + WS_B2F);
    float*  b3f = (float*)(ws + WS_B3F);

    fold_weights_kernel<<<(PREP_ELEMS + 255) / 256, 256, 0, stream>>>(
        w1, b1, g1, be1, m1, v1, w2, b2, g2, be2, m2, v2, w3, b3,
        w1b, w2b, w3b, b1f, b2f, b3f);

    hipFuncSetAttribute((const void*)episode_kernel,
                        hipFuncAttributeMaxDynamicSharedMemorySize, SMEM_BYTES);

    episode_kernel<<<NBATCH, 256, SMEM_BYTES, stream>>>(
        node, w1b, w2b, w3b, b1f, b2f, b3f, (float*)d_out);
}
